// GlobalMixtureEncoderStratifiedEOL_67551245631807
// MI455X (gfx1250) — matrix operand order repaired and checked by reading, not compiled, not run on hardware
//
#include <hip/hip_runtime.h>

// Problem constants (match reference)
#define BB   1024
#define GG   512
#define KK   128
#define EE2  128
#define KP1  129
#define KJ   (KK * KP1)   // 16512, divisible by 32
#define SPLITK 8

typedef _Float16 v16h __attribute__((ext_vector_type(16)));
typedef float    v8f  __attribute__((ext_vector_type(8)));

union FragH { v16h h; uint4 q[2]; };

struct Frags { FragH a0, a1, b0, b1, b2, b3; };

__device__ __forceinline__ void load_frags(Frags& f,
                                           const _Float16* __restrict__ Ar0,
                                           const _Float16* __restrict__ Ar1,
                                           const _Float16* __restrict__ B0,
                                           const _Float16* __restrict__ B1,
                                           const _Float16* __restrict__ B2,
                                           const _Float16* __restrict__ B3,
                                           int kk, int kA0, int kB0) {
  f.a0.q[0] = *(const uint4*)(Ar0 + kk + kA0);
  f.a0.q[1] = *(const uint4*)(Ar0 + kk + kA0 + 8);
  f.a1.q[0] = *(const uint4*)(Ar1 + kk + kA0);
  f.a1.q[1] = *(const uint4*)(Ar1 + kk + kA0 + 8);
  f.b0.q[0] = *(const uint4*)(B0 + kk + kB0);
  f.b0.q[1] = *(const uint4*)(B0 + kk + kB0 + 8);
  f.b1.q[0] = *(const uint4*)(B1 + kk + kB0);
  f.b1.q[1] = *(const uint4*)(B1 + kk + kB0 + 8);
  f.b2.q[0] = *(const uint4*)(B2 + kk + kB0);
  f.b2.q[1] = *(const uint4*)(B2 + kk + kB0 + 8);
  f.b3.q[0] = *(const uint4*)(B3 + kk + kB0);
  f.b3.q[1] = *(const uint4*)(B3 + kk + kB0 + 8);
}

__device__ __forceinline__ void mma_frags(v8f acc[2][4], const Frags& f) {
  acc[0][0] = __builtin_amdgcn_wmma_f32_16x16x32_f16(false, f.a0.h, false, f.b0.h, (short)0, acc[0][0], false, false);
  acc[1][0] = __builtin_amdgcn_wmma_f32_16x16x32_f16(false, f.a1.h, false, f.b0.h, (short)0, acc[1][0], false, false);
  acc[0][1] = __builtin_amdgcn_wmma_f32_16x16x32_f16(false, f.a0.h, false, f.b1.h, (short)0, acc[0][1], false, false);
  acc[1][1] = __builtin_amdgcn_wmma_f32_16x16x32_f16(false, f.a1.h, false, f.b1.h, (short)0, acc[1][1], false, false);
  acc[0][2] = __builtin_amdgcn_wmma_f32_16x16x32_f16(false, f.a0.h, false, f.b2.h, (short)0, acc[0][2], false, false);
  acc[1][2] = __builtin_amdgcn_wmma_f32_16x16x32_f16(false, f.a1.h, false, f.b2.h, (short)0, acc[1][2], false, false);
  acc[0][3] = __builtin_amdgcn_wmma_f32_16x16x32_f16(false, f.a0.h, false, f.b3.h, (short)0, acc[0][3], false, false);
  acc[1][3] = __builtin_amdgcn_wmma_f32_16x16x32_f16(false, f.a1.h, false, f.b3.h, (short)0, acc[1][3], false, false);
}

// ---------------------------------------------------------------------------
// Generic NT GEMM with optional split-K:
//   C_part[z][M,N] (f32) = A[M, k0:k1] (f16 row-major) x B[N, k0:k1]^T (f16 row-major)
// M%128==0, N%128==0, Kd%32==0.  Block = 256 threads = 8 waves (4 along M,
// 2 along N); each wave computes a 32x64 patch as a 2x4 grid of
// v_wmma_f32_16x16x32_f16.  Fragments are double-buffered in registers
// (software pipeline, unroll-by-2) so each 8-WMMA batch runs with the next
// batch's 12 b128 loads already in flight.
// ---------------------------------------------------------------------------
__global__ void __launch_bounds__(256)
gemm_nt_kernel(const _Float16* __restrict__ A,
               const _Float16* __restrict__ B,
               float* __restrict__ C,
               int M, int N, int Kd, int ksplit) {
  const int lane = threadIdx.x & 31;
  const int wave = threadIdx.x >> 5;
  const int wm = wave & 3;        // 4 waves along M
  const int wn = wave >> 2;       // 2 waves along N
  const int r  = lane & 15;
  const int hs = lane >> 4;       // half-select (lanes 16-31)
  const int m0 = blockIdx.x * 128 + wm * 32;
  const int n0 = blockIdx.y * 128 + wn * 64;

  const int k0 = blockIdx.z * ksplit;
  const int kend = (k0 + ksplit < Kd) ? (k0 + ksplit) : Kd;
  C += (size_t)blockIdx.z * M * N;

  const int kA0 = hs * 16;        // A lanes 16-31 hold K=16..31
  const int kB0 = hs * 16;        // B lanes 16-31 hold K=16..31

  const _Float16* Ar0 = A + (size_t)(m0 + r)      * Kd;
  const _Float16* Ar1 = A + (size_t)(m0 + 16 + r) * Kd;
  const _Float16* B0  = B + (size_t)(n0 +  0 + r) * Kd;
  const _Float16* B1  = B + (size_t)(n0 + 16 + r) * Kd;
  const _Float16* B2  = B + (size_t)(n0 + 32 + r) * Kd;
  const _Float16* B3  = B + (size_t)(n0 + 48 + r) * Kd;

  v8f acc[2][4];
#pragma unroll
  for (int mi = 0; mi < 2; ++mi)
#pragma unroll
    for (int ni = 0; ni < 4; ++ni) acc[mi][ni] = (v8f){};

  const int nsteps = (kend - k0) >> 5;   // >= 1 for all launches here
  Frags f0, f1;
  load_frags(f0, Ar0, Ar1, B0, B1, B2, B3, k0, kA0, kB0);

  int s = 0;
  for (; s + 2 <= nsteps; s += 2) {
    const int kk = k0 + s * 32;
    load_frags(f1, Ar0, Ar1, B0, B1, B2, B3, kk + 32, kA0, kB0);
    if (s + 2 < nsteps) {
      __builtin_prefetch(Ar0 + kk + 96, 0, 1);   // global_prefetch_b8, 2 steps out
      __builtin_prefetch(Ar1 + kk + 96, 0, 1);
      __builtin_prefetch(B0 + kk + 96, 0, 1);
      __builtin_prefetch(B2 + kk + 96, 0, 1);
    }
    mma_frags(acc, f0);
    if (s + 2 < nsteps)
      load_frags(f0, Ar0, Ar1, B0, B1, B2, B3, kk + 64, kA0, kB0);
    mma_frags(acc, f1);
  }
  if (s < nsteps)                 // odd tail: f0 already holds the last step
    mma_frags(acc, f0);

  // C/D layout: lane (hs,r) -> N = n0+ni*16+r ; VGPR v -> M = m0+mi*16+hs*8+v
#pragma unroll
  for (int mi = 0; mi < 2; ++mi)
#pragma unroll
    for (int ni = 0; ni < 4; ++ni) {
      const int cm = m0 + mi * 16 + hs * 8;
      const int cn = n0 + ni * 16 + r;
      float* Cp = C + (size_t)cm * N + cn;
#pragma unroll
      for (int v = 0; v < 8; ++v) Cp[(size_t)v * N] = acc[mi][ni][v];
    }
}

// ---------------------------------------------------------------------------
// Small helper kernels
// ---------------------------------------------------------------------------
__global__ void f32_to_f16(const float* __restrict__ s, _Float16* __restrict__ d, int n) {
  int i = blockIdx.x * blockDim.x + threadIdx.x;
  if (i < n) d[i] = (_Float16)s[i];
}

__global__ void transpose_f32_to_f16(const float* __restrict__ s, _Float16* __restrict__ d,
                                     int R, int C) {
  int i = blockIdx.x * blockDim.x + threadIdx.x;
  if (i < R * C) {
    int r = i / C, c = i - r * C;
    d[(size_t)c * R + r] = (_Float16)s[i];
  }
}

__global__ void col_sqnorm(const float* __restrict__ s, float* __restrict__ out, int R, int C) {
  int c = blockIdx.x * blockDim.x + threadIdx.x;
  if (c < C) {
    float a = 0.f;
    for (int r = 0; r < R; ++r) { float v = s[(size_t)r * C + c]; a += v * v; }
    out[c] = a;
  }
}

__global__ void row_sqnorm(const float* __restrict__ s, float* __restrict__ out, int R, int C) {
  int r = blockIdx.x * blockDim.x + threadIdx.x;
  if (r < R) {
    float a = 0.f;
    const float* p = s + (size_t)r * C;
    for (int c = 0; c < C; ++c) { float v = p[c]; a += v * v; }
    out[r] = a;
  }
}

// Per-code stats: t2[k,j] = sum_g wimg[g,k]*wimg2[g,k,j];  nw2[k,j] = sum_g wimg2[g,k,j]^2
__global__ void wimg2_stats(const float* __restrict__ wimg, const float* __restrict__ wimg2,
                            float* __restrict__ t2, float* __restrict__ nw2) {
  __shared__ float wcol[GG];
  const int k = blockIdx.x;
  for (int g = threadIdx.x; g < GG; g += blockDim.x) wcol[g] = wimg[(size_t)g * KK + k];
  __syncthreads();
  const int j = threadIdx.x;
  if (j < KP1) {
    float at = 0.f, an = 0.f;
    const float* p = wimg2 + (size_t)k * KP1 + j;
    for (int g = 0; g < GG; ++g) {
      float w = p[(size_t)g * KJ];
      at += wcol[g] * w;
      an += w * w;
    }
    t2[k * KP1 + j] = at;
    nw2[k * KP1 + j] = an;
  }
}

// softmax over K of (2*S - nrm[k]) * invD  -> f16 output only (feeds next GEMM)
__global__ void softmax_enc1(const float* __restrict__ S, const float* __restrict__ nrm,
                             _Float16* __restrict__ outh, float invD) {
  __shared__ float red[KK];
  const int b = blockIdx.x, k = threadIdx.x;
  float l = (2.f * S[(size_t)b * KK + k] - nrm[k]) * invD;
  red[k] = l; __syncthreads();
  for (int s = KK / 2; s > 0; s >>= 1) { if (k < s) red[k] = fmaxf(red[k], red[k + s]); __syncthreads(); }
  float m = red[0]; __syncthreads();
  float e = __expf(l - m);
  red[k] = e; __syncthreads();
  for (int s = KK / 2; s > 0; s >>= 1) { if (k < s) red[k] += red[k + s]; __syncthreads(); }
  outh[(size_t)b * KK + k] = (_Float16)(e / red[0]);
}

// softmax + argmax: xp_d (f32 & f16), idx
__global__ void softmax_enc2(const float* __restrict__ S, const float* __restrict__ nrm,
                             float* __restrict__ xpd, _Float16* __restrict__ xpdh,
                             int* __restrict__ idx, float invD) {
  __shared__ float rv[KK];
  __shared__ int   ri[KK];
  const int b = blockIdx.x, k = threadIdx.x;
  float l = (2.f * S[(size_t)b * KK + k] - nrm[k]) * invD;
  rv[k] = l; ri[k] = k; __syncthreads();
  for (int s = KK / 2; s > 0; s >>= 1) {
    if (k < s && rv[k + s] > rv[k]) { rv[k] = rv[k + s]; ri[k] = ri[k + s]; }
    __syncthreads();
  }
  float m = rv[0]; int im = ri[0]; __syncthreads();
  float e = __expf(l - m);
  rv[k] = e; __syncthreads();
  for (int s = KK / 2; s > 0; s >>= 1) { if (k < s) rv[k] += rv[k + s]; __syncthreads(); }
  float p = e / rv[0];
  xpd[(size_t)b * KK + k] = p;
  xpdh[(size_t)b * KK + k] = (_Float16)p;
  if (k == 0) idx[b] = im;
}

// Per-batch stratified chain: P -> xp2 -> lat2 -> xp2_d -> Aout (outer product, f16)
__global__ void per_b_kernel(const float* __restrict__ images, const float* __restrict__ wimg2,
                             const float* __restrict__ t2, const float* __restrict__ nw2,
                             const float* __restrict__ wrec2, const float* __restrict__ nrec2,
                             const float* __restrict__ xpd, const int* __restrict__ idx,
                             _Float16* __restrict__ Aout) {
  __shared__ float img[GG];
  __shared__ float xp2[KP1];
  __shared__ float lat2[EE2];
  __shared__ float xp2d[KP1];
  __shared__ float xpk[KK];
  __shared__ float red[256];
  const int b = blockIdx.x, tid = threadIdx.x;
  const int kidx = idx[b];

  for (int g = tid; g < GG; g += 256) img[g] = images[(size_t)b * GG + g];
  __syncthreads();

  // P[j] and logits l3
  float l3 = -3.0e38f;
  if (tid < KP1) {
    float P = 0.f;
    const float* p = wimg2 + (size_t)kidx * KP1 + tid;
    for (int g = 0; g < GG; ++g) P += img[g] * p[(size_t)g * KJ];
    l3 = (2.f * (P - t2[kidx * KP1 + tid]) - nw2[kidx * KP1 + tid]) * (1.f / (float)GG);
  }
  // softmax over 129 -> xp2
  red[tid] = l3; __syncthreads();
  for (int s = 128; s > 0; s >>= 1) { if (tid < s) red[tid] = fmaxf(red[tid], red[tid + s]); __syncthreads(); }
  float m3 = red[0]; __syncthreads();
  float e3 = (tid < KP1) ? __expf(l3 - m3) : 0.f;
  red[tid] = e3; __syncthreads();
  for (int s = 128; s > 0; s >>= 1) { if (tid < s) red[tid] += red[tid + s]; __syncthreads(); }
  float inv3 = 1.f / red[0];
  if (tid < KP1) xp2[tid] = e3 * inv3;
  __syncthreads();

  // lat2 = xp2 @ wrec2  [129 x 128]
  if (tid < EE2) {
    float a = 0.f;
    for (int j = 0; j < KP1; ++j) a += xp2[j] * wrec2[(size_t)j * EE2 + tid];
    lat2[tid] = a;
  }
  __syncthreads();

  // logits l4 = (2*lat2.wrec2[j] - ||wrec2[j]||^2)/E2 -> softmax -> xp2_d
  float l4 = -3.0e38f;
  if (tid < KP1) {
    float s3 = 0.f;
    const float* wr = wrec2 + (size_t)tid * EE2;
    for (int e = 0; e < EE2; ++e) s3 += lat2[e] * wr[e];
    l4 = (2.f * s3 - nrec2[tid]) * (1.f / (float)EE2);
  }
  __syncthreads();
  red[tid] = l4; __syncthreads();
  for (int s = 128; s > 0; s >>= 1) { if (tid < s) red[tid] = fmaxf(red[tid], red[tid + s]); __syncthreads(); }
  float m4 = red[0]; __syncthreads();
  float e4 = (tid < KP1) ? __expf(l4 - m4) : 0.f;
  red[tid] = e4; __syncthreads();
  for (int s = 128; s > 0; s >>= 1) { if (tid < s) red[tid] += red[tid + s]; __syncthreads(); }
  float inv4 = 1.f / red[0];
  if (tid < KP1) xp2d[tid] = e4 * inv4;
  if (tid < KK) xpk[tid] = xpd[(size_t)b * KK + tid];
  __syncthreads();

  // Aout[b, k*129+j] = xp_d[k] * xp2_d[j]  (f16, feeds big WMMA GEMM)
  _Float16* Ab = Aout + (size_t)b * KJ;
  for (int i = tid; i < KJ; i += 256) {
    int k = i / KP1, j = i - k * KP1;
    Ab[i] = (_Float16)(xpk[k] * xp2d[j]);
  }
}

// Fused split-K reduction + correction + MSE loss:
// loss[b] = mean_g ( corr[b,g] + sum_z parts[z][b,g] - images[b,g] )^2
__global__ void loss_reduce_kernel(const float* __restrict__ corr,
                                   const float* __restrict__ parts,
                                   const float* __restrict__ images,
                                   float* __restrict__ out) {
  __shared__ float red[256];
  const int b = blockIdx.x, tid = threadIdx.x;
  float a = 0.f;
  for (int g = tid; g < GG; g += 256) {
    float x = corr[(size_t)b * GG + g];
#pragma unroll
    for (int z = 0; z < SPLITK; ++z) x += parts[(size_t)z * BB * GG + (size_t)b * GG + g];
    float d = x - images[(size_t)b * GG + g];
    a += d * d;
  }
  red[tid] = a; __syncthreads();
  for (int s = 128; s > 0; s >>= 1) { if (tid < s) red[tid] += red[tid + s]; __syncthreads(); }
  if (tid == 0) out[b] = red[0] * (1.f / (float)GG);
}

// ---------------------------------------------------------------------------
extern "C" void kernel_launch(void* const* d_in, const int* in_sizes, int n_in,
                              void* d_out, int out_size, void* d_ws, size_t ws_size,
                              hipStream_t stream) {
  (void)in_sizes; (void)n_in; (void)out_size; (void)ws_size;
  const float* images = (const float*)d_in[0];   // [B,G]
  const float* wimg   = (const float*)d_in[1];   // [G,K]
  const float* wrec   = (const float*)d_in[2];   // [K,E2]
  const float* wrec2  = (const float*)d_in[3];   // [K+1,E2]
  const float* wimg2  = (const float*)d_in[4];   // [G,K,K+1]
  float* loss = (float*)d_out;                   // [B]

  char* ws = (char*)d_ws;
  size_t off = 0;
  auto alloc = [&](size_t bytes) -> void* {
    void* p = ws + off;
    off += (bytes + 255) & ~(size_t)255;
    return p;
  };

  _Float16* imagesh = (_Float16*)alloc((size_t)BB * GG * 2);
  _Float16* wimg_h  = (_Float16*)alloc((size_t)GG * KK * 2);
  _Float16* wimgT_h = (_Float16*)alloc((size_t)KK * GG * 2);
  _Float16* wrec_h  = (_Float16*)alloc((size_t)KK * EE2 * 2);
  _Float16* wrecT_h = (_Float16*)alloc((size_t)EE2 * KK * 2);
  _Float16* wimg2h  = (_Float16*)alloc((size_t)GG * KJ * 2);
  _Float16* Aout    = (_Float16*)alloc((size_t)BB * KJ * 2);
  float* S1    = (float*)alloc((size_t)BB * KK * 4);
  _Float16* xp_h  = (_Float16*)alloc((size_t)BB * KK * 2);
  float* lat   = (float*)alloc((size_t)BB * EE2 * 4);
  _Float16* lat_h = (_Float16*)alloc((size_t)BB * EE2 * 2);
  float* S2    = (float*)alloc((size_t)BB * KK * 4);
  float* xp_d  = (float*)alloc((size_t)BB * KK * 4);
  _Float16* xp_dh = (_Float16*)alloc((size_t)BB * KK * 2);
  int*   idx   = (int*)alloc((size_t)BB * 4);
  float* n_w   = (float*)alloc(KK * 4);
  float* nrec  = (float*)alloc(KK * 4);
  float* nrec2 = (float*)alloc(KP1 * 4);
  float* t2    = (float*)alloc((size_t)KK * KP1 * 4);
  float* nw2   = (float*)alloc((size_t)KK * KP1 * 4);
  float* corr  = (float*)alloc((size_t)BB * GG * 4);
  float* parts = (float*)alloc((size_t)SPLITK * BB * GG * 4);

  // --- conversions & precomputed stats ---
  f32_to_f16<<<(BB * GG + 255) / 256, 256, 0, stream>>>(images, imagesh, BB * GG);
  f32_to_f16<<<(GG * KK + 255) / 256, 256, 0, stream>>>(wimg, wimg_h, GG * KK);
  f32_to_f16<<<(KK * EE2 + 255) / 256, 256, 0, stream>>>(wrec, wrec_h, KK * EE2);
  f32_to_f16<<<(GG * KJ + 255) / 256, 256, 0, stream>>>(wimg2, wimg2h, GG * KJ);
  transpose_f32_to_f16<<<(GG * KK + 255) / 256, 256, 0, stream>>>(wimg, wimgT_h, GG, KK);
  transpose_f32_to_f16<<<(KK * EE2 + 255) / 256, 256, 0, stream>>>(wrec, wrecT_h, KK, EE2);
  col_sqnorm<<<1, 128, 0, stream>>>(wimg, n_w, GG, KK);
  row_sqnorm<<<1, 128, 0, stream>>>(wrec, nrec, KK, EE2);
  row_sqnorm<<<2, 128, 0, stream>>>(wrec2, nrec2, KP1, EE2);
  wimg2_stats<<<KK, 256, 0, stream>>>(wimg, wimg2, t2, nw2);

  // --- encode (single-slice GEMMs: ksplit == Kd) ---
  gemm_nt_kernel<<<dim3(BB / 128, KK / 128, 1), 256, 0, stream>>>(imagesh, wimgT_h, S1, BB, KK, GG, GG);
  softmax_enc1<<<BB, 128, 0, stream>>>(S1, n_w, xp_h, 1.f / (float)GG);
  gemm_nt_kernel<<<dim3(BB / 128, EE2 / 128, 1), 256, 0, stream>>>(xp_h, wrecT_h, lat, BB, EE2, KK, KK);
  f32_to_f16<<<(BB * EE2 + 255) / 256, 256, 0, stream>>>(lat, lat_h, BB * EE2);
  gemm_nt_kernel<<<dim3(BB / 128, KK / 128, 1), 256, 0, stream>>>(lat_h, wrec_h, S2, BB, KK, EE2, EE2);
  softmax_enc2<<<BB, 128, 0, stream>>>(S2, nrec, xp_d, xp_dh, idx, 1.f / (float)EE2);

  // --- stratified second stage (per-b) -> Aout outer products ---
  per_b_kernel<<<BB, 256, 0, stream>>>(images, wimg2, t2, nw2, wrec2, nrec2, xp_d, idx, Aout);

  // --- decode ---
  // correction term: corr = xp_d @ wimg^T  (Kd = 128)
  gemm_nt_kernel<<<dim3(BB / 128, GG / 128, 1), 256, 0, stream>>>(xp_dh, wimg_h, corr, BB, GG, KK, KK);
  // big einsum: parts[z] = Aout @ wimg2^T over K-slice z  (split-K = 8, 2080 each)
  const int ksplit = ((KJ / 32 + SPLITK - 1) / SPLITK) * 32;   // 2080
  gemm_nt_kernel<<<dim3(BB / 128, GG / 128, SPLITK), 256, 0, stream>>>(Aout, wimg2h, parts, BB, GG, KJ, ksplit);

  // --- fused split-K reduction + loss ---
  loss_reduce_kernel<<<BB, 256, 0, stream>>>(corr, parts, images, loss);
}
